// LigerFusedLinearCPOLoss_18391049961623
// MI455X (gfx1250) — matrix operand order repaired and checked by reading, not compiled, not run on hardware
//
#include <hip/hip_runtime.h>
#include <math.h>

typedef __attribute__((ext_vector_type(16))) __bf16 v16bf;
typedef __attribute__((ext_vector_type(2)))  __bf16 v2bf;
typedef __attribute__((ext_vector_type(8)))  float  v8f;

#define IGNORE_INDEX (-100)
#define BETA  0.1f
#define ALPHA 1.0f

#define Hdim 4096
#define Vdim 32000
#define BT   8192
#define MT   64          // tokens per workgroup
#define NT   256         // vocab per chunk (4 wave-cols * 4 frags * 16)
#define KT   32
#define ASTRIDE 40       // bf16 elements per LDS row (pad 8 to skew banks)
#define BSTRIDE 40
#define NKS (Hdim / KT)  // 128
#define NCH (Vdim / NT)  // 125

// ---- bf16 packing: prefer hardware v_cvt_pk_bf16_f32, else RNE + v_perm_b32 ----
__device__ __forceinline__ unsigned pack_bf16(float lo, float hi) {
#if defined(__gfx1250__) && __has_builtin(__builtin_amdgcn_cvt_pk_bf16_f32)
  v2bf r = __builtin_amdgcn_cvt_pk_bf16_f32(lo, hi);
  return __builtin_bit_cast(unsigned, r);
#else
  unsigned a = __builtin_bit_cast(unsigned, lo);
  unsigned b = __builtin_bit_cast(unsigned, hi);
  a += 0x7FFFu + ((a >> 16) & 1u);   // round-to-nearest-even
  b += 0x7FFFu + ((b >> 16) & 1u);
  // result bytes = {a.b2, a.b3, b.b2, b.b3}
  return __builtin_amdgcn_perm(b, a, 0x07060302u);
#endif
}

// ---- CDNA5 async global->LDS copy (16B per lane), tracked by ASYNCcnt ----
__device__ __forceinline__ void async_ld16(unsigned lds_byte_addr, const void* gaddr) {
  asm volatile("global_load_async_to_lds_b128 %0, %1, off"
               :: "v"(lds_byte_addr), "v"(gaddr) : "memory");
}
__device__ __forceinline__ void wait_async0() {
  asm volatile("s_wait_asynccnt 0x0" ::: "memory");
}

struct Frag32 { uint4 x, y; };

// A frag: lane holds row m=lane%16; elems 0..7 = K g*8.., elems 8..15 = K 16+g*8..
__device__ __forceinline__ v16bf ld_fragA(const unsigned short* base) {
  Frag32 f;
  f.x = *reinterpret_cast<const uint4*>(base);
  f.y = *reinterpret_cast<const uint4*>(base + 16);
  return __builtin_bit_cast(v16bf, f);
}
// B frag: lane holds col n=lane%16; elems 0..7 = K g*8.., elems 8..15 = K 16+g*8..
__device__ __forceinline__ v16bf ld_fragB(const unsigned short* base) {
  Frag32 f;
  f.x = *reinterpret_cast<const uint4*>(base);
  f.y = *reinterpret_cast<const uint4*>(base + 16);
  return __builtin_bit_cast(v16bf, f);
}

// ---- one-time fp32 -> bf16 conversion (pre-convert path) ----
__global__ void cvt_f32_bf16(const float* __restrict__ src,
                             unsigned short* __restrict__ dst, long long n8) {
  long long i = (long long)blockIdx.x * blockDim.x + threadIdx.x;
  const long long stride = (long long)gridDim.x * blockDim.x;
  for (; i < n8; i += stride) {
    const float4 a = *reinterpret_cast<const float4*>(src + i * 8);
    const float4 b = *reinterpret_cast<const float4*>(src + i * 8 + 4);
    uint4 o;
    o.x = pack_bf16(a.x, a.y); o.y = pack_bf16(a.z, a.w);
    o.z = pack_bf16(b.x, b.y); o.w = pack_bf16(b.z, b.w);
    *reinterpret_cast<uint4*>(dst + i * 8) = o;
  }
}

template<bool PRE>
__global__ __launch_bounds__(256, 1)
void fused_linear_cpo_logp(const float* __restrict__ W,
                           const float* __restrict__ X,
                           const unsigned short* __restrict__ Wb,
                           const unsigned short* __restrict__ Xb,
                           const int*   __restrict__ T,
                           const float* __restrict__ bias,
                           float* __restrict__ logp_out)
{
  __shared__ unsigned short As[2][MT * ASTRIDE];
  __shared__ unsigned short Bs[2][NT * BSTRIDE];
  __shared__ float red_max[2][4][32];
  __shared__ float red_sum[2][4][32];
  __shared__ float run_max[MT];
  __shared__ float run_sum[MT];
  __shared__ float tgt_raw[MT];
  __shared__ int   tgt_t[MT];

  const int tid  = threadIdx.x;
  const int lane = tid & 31;
  const int wave = tid >> 5;
  const int wm   = wave >> 2;   // 0..1  (32 token rows each)
  const int wn   = wave & 3;    // 0..3  (64 vocab cols each)
  const int n16  = lane & 15;
  const int g    = lane >> 4;
  const int mbase = blockIdx.x * MT;

  if (tid < MT) {
    run_max[tid] = -INFINITY;
    run_sum[tid] = 0.f;
    tgt_raw[tid] = 0.f;
    tgt_t[tid]   = T[mbase + tid];
  }

  const int arow = tid >> 2;   // 0..63
  const int akc  = tid & 3;    // 0..3 -> 8 K elements each
  const float*          aptrF = X  + (size_t)(mbase + arow) * Hdim + akc * 8;
  const unsigned short* aptrH = Xb + (size_t)(mbase + arow) * Hdim + akc * 8;

  // LDS byte addresses of this thread's store slots (generic ptr low 32b = LDS offset)
  const unsigned ldsA0 = (unsigned)(uintptr_t)&As[0][arow * ASTRIDE + akc * 8];
  const unsigned ldsA1 = (unsigned)(uintptr_t)&As[1][arow * ASTRIDE + akc * 8];
  const unsigned ldsB0 = (unsigned)(uintptr_t)&Bs[0][tid * BSTRIDE];
  const unsigned ldsB1 = (unsigned)(uintptr_t)&Bs[1][tid * BSTRIDE];

  __syncthreads();

  for (int vc = 0; vc < NCH; ++vc) {
    const int vbase = vc * NT;
    const float*          bptrF = W  + (size_t)(vbase + tid) * Hdim;
    const unsigned short* bptrH = Wb + (size_t)(vbase + tid) * Hdim;

    v8f acc[2][4];
    for (int fm = 0; fm < 2; ++fm)
      for (int fn = 0; fn < 4; ++fn)
        for (int j = 0; j < 8; ++j) acc[fm][fn][j] = 0.f;

    float4 sa0, sa1, sb[8];   // raw-fp32 staging (fallback path only)

    auto issue_async = [&](int kidx, int buf) {
      const unsigned la = buf ? ldsA1 : ldsA0;
      const unsigned lb = buf ? ldsB1 : ldsB0;
      async_ld16(la, aptrH + kidx * KT);
      #pragma unroll
      for (int i = 0; i < 4; ++i)
        async_ld16(lb + i * 16, bptrH + kidx * KT + i * 8);
    };
    auto gload_raw = [&](int kidx) {
      const float* ap = aptrF + kidx * KT;
      sa0 = *reinterpret_cast<const float4*>(ap);
      sa1 = *reinterpret_cast<const float4*>(ap + 4);
      const float* bp = bptrF + kidx * KT;
      #pragma unroll
      for (int i = 0; i < 8; ++i)
        sb[i] = *reinterpret_cast<const float4*>(bp + i * 4);
    };
    auto cvt_store = [&](int buf) {    // runs AFTER the WMMA block
      *reinterpret_cast<uint4*>(&As[buf][arow * ASTRIDE + akc * 8]) =
          make_uint4(pack_bf16(sa0.x, sa0.y), pack_bf16(sa0.z, sa0.w),
                     pack_bf16(sa1.x, sa1.y), pack_bf16(sa1.z, sa1.w));
      uint4* pb = reinterpret_cast<uint4*>(&Bs[buf][tid * BSTRIDE]);
      #pragma unroll
      for (int i = 0; i < 4; ++i)
        pb[i] = make_uint4(pack_bf16(sb[2*i].x,   sb[2*i].y),
                           pack_bf16(sb[2*i].z,   sb[2*i].w),
                           pack_bf16(sb[2*i+1].x, sb[2*i+1].y),
                           pack_bf16(sb[2*i+1].z, sb[2*i+1].w));
    };

    // ---- prologue: k-step 0 -> buf 0 ----
    if constexpr (PRE) {
      issue_async(0, 0);
      wait_async0();
    } else {
      gload_raw(0);
      cvt_store(0);
    }
    __syncthreads();

    for (int ks = 0; ks < NKS; ++ks) {
      const int cb = ks & 1;
      const bool more = (ks + 1) < NKS;
      if (more) {                      // start next k-step's global traffic early
        if constexpr (PRE) issue_async(ks + 1, cb ^ 1);
        else               gload_raw(ks + 1);
      }
      // ---- compute from buf cb: 12 ds_load_b128 + 8 WMMA ----
      v16bf bfr[4];
      #pragma unroll
      for (int fn = 0; fn < 4; ++fn)
        bfr[fn] = ld_fragB(&Bs[cb][(wn*64 + fn*16 + n16) * BSTRIDE + g*8]);
      #pragma unroll
      for (int fm = 0; fm < 2; ++fm) {
        const v16bf afr = ld_fragA(&As[cb][(wm*32 + fm*16 + n16) * ASTRIDE + g*8]);
        #pragma unroll
        for (int fn = 0; fn < 4; ++fn)
          acc[fm][fn] = __builtin_amdgcn_wmma_f32_16x16x32_bf16(
              false, afr, false, bfr[fn], (short)0, acc[fm][fn], false, false);
      }
      if (more) {
        if constexpr (PRE) wait_async0();   // asyncs had the WMMA block to land
        else               cvt_store(cb ^ 1);  // loadcnt wait sits AFTER the WMMAs
      }
      __syncthreads();
    }

    // ---- bias + streaming log-softmax update for this vocab chunk ----
    // C/D layout: vgpr j, lane -> row fm*16 + g*8 + j, col n16
    for (int fn = 0; fn < 4; ++fn) {
      const float bv = bias[vbase + wn*64 + fn*16 + n16];
      for (int fm = 0; fm < 2; ++fm)
        for (int j = 0; j < 8; ++j) acc[fm][fn][j] += bv;
    }
    for (int fm = 0; fm < 2; ++fm) {
      float rmax[8];
      for (int j = 0; j < 8; ++j) rmax[j] = -INFINITY;
      for (int fn = 0; fn < 4; ++fn)
        for (int j = 0; j < 8; ++j) rmax[j] = fmaxf(rmax[j], acc[fm][fn][j]);
      for (int j = 0; j < 8; ++j)
        for (int off = 8; off >= 1; off >>= 1)
          rmax[j] = fmaxf(rmax[j], __shfl_xor(rmax[j], off, 16));
      if (n16 == 0)
        for (int j = 0; j < 8; ++j)
          red_max[wm][wn][fm*16 + g*8 + j] = rmax[j];
    }
    __syncthreads();
    for (int fm = 0; fm < 2; ++fm) {
      float nm[8], s[8];
      for (int j = 0; j < 8; ++j) {
        const int r = fm*16 + g*8 + j;
        float mx = run_max[wm*32 + r];
        for (int wv = 0; wv < 4; ++wv) mx = fmaxf(mx, red_max[wm][wv][r]);
        nm[j] = mx; s[j] = 0.f;
      }
      for (int fn = 0; fn < 4; ++fn)
        for (int j = 0; j < 8; ++j) s[j] += __expf(acc[fm][fn][j] - nm[j]);
      for (int j = 0; j < 8; ++j)
        for (int off = 8; off >= 1; off >>= 1)
          s[j] += __shfl_xor(s[j], off, 16);
      if (n16 == 0)
        for (int j = 0; j < 8; ++j)
          red_sum[wm][wn][fm*16 + g*8 + j] = s[j];
      for (int j = 0; j < 8; ++j) {           // target-logit extraction
        const int r  = fm*16 + g*8 + j;
        const int tv = tgt_t[wm*32 + r];
        for (int fn = 0; fn < 4; ++fn)
          if (vbase + wn*64 + fn*16 + n16 == tv)
            tgt_raw[wm*32 + r] = acc[fm][fn][j];
      }
    }
    __syncthreads();
    if (wn == 0) {                            // online rescale, 1 lane per row
      const int r = lane;
      const int x = wm*32 + r;
      float mx = run_max[x];
      for (int wv = 0; wv < 4; ++wv) mx = fmaxf(mx, red_max[wm][wv][r]);
      float sa = run_sum[x] * __expf(run_max[x] - mx);
      for (int wv = 0; wv < 4; ++wv) sa += red_sum[wm][wv][r];
      run_max[x] = mx; run_sum[x] = sa;
    }
    __syncthreads();
  }

  if (tid < MT) {
    float lp = tgt_raw[tid] - run_max[tid] - __logf(run_sum[tid]);
    if (tgt_t[tid] == IGNORE_INDEX) lp = 0.f;
    logp_out[mbase + tid] = lp;
  }
}

__global__ void cpo_finalize(const float* __restrict__ logp,
                             const int* __restrict__ T,
                             float* __restrict__ out)
{
  __shared__ float red[1024];
  __shared__ float seq[8];
  const int tid = threadIdx.x;
  float chosen_cnt = 0.f;
  for (int s = 0; s < 8; ++s) {
    const int t = s * 1024 + tid;
    red[tid] = logp[t];
    if (s < 4 && T[t] != IGNORE_INDEX) chosen_cnt += 1.f;
    __syncthreads();
    for (int off = 512; off > 0; off >>= 1) {
      if (tid < off) red[tid] += red[tid + off];
      __syncthreads();
    }
    if (tid == 0) seq[s] = red[0];
    __syncthreads();
  }
  red[tid] = chosen_cnt;
  __syncthreads();
  for (int off = 512; off > 0; off >>= 1) {
    if (tid < off) red[tid] += red[tid + off];
    __syncthreads();
  }
  if (tid == 0) {
    const float n_chosen   = red[0];
    const float chosen_sum = seq[0] + seq[1] + seq[2] + seq[3];
    const float nll = -chosen_sum / n_chosen;
    float pref = 0.f;
    for (int i = 0; i < 4; ++i) {
      const float d  = BETA * (seq[i] - seq[4 + i]);
      const float ls = fminf(d, 0.f) - log1pf(__expf(-fabsf(d)));  // log_sigmoid
      pref += -ls;
    }
    pref *= 0.25f;   // / half
    out[0] = ALPHA * nll + pref;
  }
}

extern "C" void kernel_launch(void* const* d_in, const int* in_sizes, int n_in,
                              void* d_out, int out_size, void* d_ws, size_t ws_size,
                              hipStream_t stream) {
  const float* W    = (const float*)d_in[0];   // lin_weight (V,H)
  const float* X    = (const float*)d_in[1];   // _input (B,T,H)
  const int*   T    = (const int*)d_in[2];     // target (B,T)
  const float* bias = (const float*)d_in[3];   // bias (V)

  const size_t needW = (size_t)Vdim * Hdim * sizeof(unsigned short); // 262 MB
  const size_t needX = (size_t)BT   * Hdim * sizeof(unsigned short); //  67 MB
  const size_t needL = (size_t)BT * sizeof(float);

  if (ws_size >= needW + needX + needL) {
    // Pre-convert operands to bf16 once (~1 GB streamed ≈ 43 us @ 23.3 TB/s),
    // then the GEMM hot loop uses async-to-LDS bf16 loads: half the bytes, no VALU cvt.
    unsigned short* Wb = (unsigned short*)d_ws;
    unsigned short* Xb = (unsigned short*)((char*)d_ws + needW);
    float* logp        = (float*)((char*)d_ws + needW + needX);
    cvt_f32_bf16<<<dim3(4096), dim3(256), 0, stream>>>(W, Wb, (long long)Vdim * Hdim / 8);
    cvt_f32_bf16<<<dim3(2048), dim3(256), 0, stream>>>(X, Xb, (long long)BT   * Hdim / 8);
    fused_linear_cpo_logp<true><<<dim3(BT / MT), dim3(256), 0, stream>>>(
        W, X, Wb, Xb, T, bias, logp);
    cpo_finalize<<<dim3(1), dim3(1024), 0, stream>>>(logp, T, (float*)d_out);
  } else {
    float* logp = (float*)d_ws;
    fused_linear_cpo_logp<false><<<dim3(BT / MT), dim3(256), 0, stream>>>(
        W, X, nullptr, nullptr, T, bias, logp);
    cpo_finalize<<<dim3(1), dim3(1024), 0, stream>>>(logp, T, (float*)d_out);
  }
}
